// LISTA_c_60043642798863
// MI455X (gfx1250) — matrix operand order repaired and checked by reading, not compiled, not run on hardware
//
#include <hip/hip_runtime.h>
#include <hip/hip_bf16.h>

// ---------------------------------------------------------------------------
// LISTA (complex) on MI455X / gfx1250.
// Compute-bound: ~360 GFLOP total, ~200MB traffic -> run the per-iteration
// cmul(B, x) as batched bf16 WMMA GEMMs (f32 accumulate), 64 samples / block.
// ---------------------------------------------------------------------------

typedef __attribute__((ext_vector_type(16))) __bf16        v16bf;
typedef __attribute__((ext_vector_type(8)))  float         v8f;
typedef __attribute__((ext_vector_type(4)))  unsigned int  u32x4;
typedef __attribute__((ext_vector_type(8)))  unsigned int  u32x8;

#define MDIM   256
#define NDIM   128
#define TITERS 10
#define NB     64      // samples per workgroup tile
#define LDY    136     // padded LDS leading dim for Y_t (128 + 8) -> conflict-free
#define LDX    264     // padded LDS leading dim for X_t / Ay_t (256 + 8)

// LDS offsets (in ushort units)
#define OFF_YRE   0
#define OFF_YIM   8704        // 64*136
#define OFF_AYRE  17408
#define OFF_AYIM  34304       // +64*264
#define OFF_X0RE  51200
#define OFF_X0IM  68096
#define OFF_X1RE  84992
#define OFF_X1IM  101888
#define LDS_USHORTS 118784    // 237568 bytes

__device__ __forceinline__ unsigned short f2bf(float f) {
  unsigned int u = __builtin_bit_cast(unsigned int, f);
  u += 0x7FFFu + ((u >> 16) & 1u);            // round-to-nearest-even
  return (unsigned short)(u >> 16);
}

__device__ __forceinline__ v16bf mkfrag(u32x4 lo, u32x4 hi) {
  u32x8 u;
  u[0] = lo[0]; u[1] = lo[1]; u[2] = lo[2]; u[3] = lo[3];
  u[4] = hi[0]; u[5] = hi[1]; u[6] = hi[2]; u[7] = hi[3];
  return __builtin_bit_cast(v16bf, u);
}

__device__ __forceinline__ v16bf negfrag(v16bf a) {      // flip both packed bf16 signs
  u32x8 u = __builtin_bit_cast(u32x8, a);
  u = u ^ 0x80008000u;
  return __builtin_bit_cast(v16bf, u);
}

// A-operand 16x32 bf16 (ISA 7.12.2): lane row = m0+(lane&15);
// elems 0..7 -> K = k0 + 8*hl + e ; elems 8..15 -> K = k0 + 16 + 8*hl + (e-8)
__device__ __forceinline__ v16bf loadA(const unsigned short* base, int ld,
                                       int row, int k0, int hl) {
  const unsigned short* p = base + row * ld + k0 + hl * 8;
  u32x4 lo = *(const u32x4*)p;
  u32x4 hi = *(const u32x4*)(p + 16);   // +32 bytes -> K+16 block
  return mkfrag(lo, hi);
}

// B-operand 32x16 bf16: lane col = n0+(lane&15); elems 0..7 -> K = k0+8*hl+e ; elems 8..15 -> K = k0+16+8*hl+(e-8)
__device__ __forceinline__ v16bf loadB(const unsigned short* base, int ld,
                                       int c, int k0, int hl) {
  const unsigned short* p = base + c * ld + k0 + hl * 8;
  u32x4 lo = *(const u32x4*)p;
  u32x4 hi = *(const u32x4*)(p + 16);   // +32 bytes -> K+16 block
  return mkfrag(lo, hi);
}

__device__ __forceinline__ v8f wmma_bf16(v16bf a, v16bf b, v8f c) {
  return __builtin_amdgcn_wmma_f32_16x16x32_bf16(
      /*neg_a=*/false, a, /*neg_b=*/false, b,
      /*c_mod=*/(short)0, c, /*reuse_a=*/false, /*reuse_b=*/false);
}

__device__ __forceinline__ void unpack8(u32x4 d, float* f) {
#pragma unroll
  for (int i = 0; i < 4; ++i) {
    f[2 * i]     = __builtin_bit_cast(float, d[i] << 16);
    f[2 * i + 1] = __builtin_bit_cast(float, d[i] & 0xFFFF0000u);
  }
}

__device__ __forceinline__ u32x4 pack8(const float* f) {
  u32x4 d;
#pragma unroll
  for (int i = 0; i < 4; ++i)
    d[i] = (unsigned)f2bf(f[2 * i]) | ((unsigned)f2bf(f[2 * i + 1]) << 16);
  return d;
}

__device__ __forceinline__ float shrinkf(float x, float eta) {
  // eta * softshrink(x/eta, 1.0) == sign(x) * max(|x| - eta, 0)
  float s = fabsf(x) - eta;
  return s > 0.f ? copysignf(s, x) : 0.f;
}

// ---------------------------------------------------------------------------
// Kernel 0: fp32 weights -> bf16 workspace.
// ws layout (ushort): [0..65535]=B_re, [65536..131071]=B_im,
//                     [131072..163839]=A_re, [163840..196607]=A_im
// ---------------------------------------------------------------------------
__global__ void convert_weights(const float* __restrict__ A,
                                const float* __restrict__ Bm,
                                unsigned short* __restrict__ wbf) {
  int i = blockIdx.x * blockDim.x + threadIdx.x;
  if (i < 2 * MDIM * MDIM) wbf[i] = f2bf(Bm[i]);
  if (i < 2 * MDIM * NDIM) wbf[2 * MDIM * MDIM + i] = f2bf(A[i]);
}

// ---------------------------------------------------------------------------
// Main kernel: 512 threads = 16 waves; wave w owns m-tile w (16 rows),
// iterates 4 n-tiles (64 samples). Dynamic LDS holds Y_t, Ay_t, double-
// buffered X_t, all transposed [sample][k] bf16 with padded leading dims.
// 237KB LDS -> 1 WG per WGP, so allow max VGPRs (no spills) via
// __launch_bounds__(512, 1); k-loops capped at unroll 2 to bound register
// pressure (32 WMMAs in flight is enough ILP).
// ---------------------------------------------------------------------------
__global__ void __launch_bounds__(512, 1)
lista_wmma_kernel(const float* __restrict__ y,
                  const unsigned short* __restrict__ wbf,
                  const float* __restrict__ etas,
                  const float* __restrict__ gammas,
                  float* __restrict__ out) {
  extern __shared__ unsigned short smem[];
  const int tid  = threadIdx.x;
  const int lane = tid & 31;
  const int w    = tid >> 5;       // wave id == m-tile id (0..15)
  const int hl   = lane >> 4;      // half-wave selector
  const int col  = lane & 15;      // row (A) / column (B,C,D) within tile
  const int b0   = blockIdx.x * NB;
  const int m0   = w * 16;
  const int mrow = m0 + 8 * hl;    // C/D: VGPR v -> m = mrow + v

  const unsigned short* Bw_re = wbf;
  const unsigned short* Bw_im = wbf + MDIM * MDIM;
  const unsigned short* Aw_re = wbf + 2 * MDIM * MDIM;
  const unsigned short* Aw_im = wbf + 2 * MDIM * MDIM + MDIM * NDIM;

  unsigned short* Yre  = smem + OFF_YRE;
  unsigned short* Yim  = smem + OFF_YIM;
  unsigned short* AYre = smem + OFF_AYRE;
  unsigned short* AYim = smem + OFF_AYIM;

  // ---- Stage 0: stage y tile into LDS, split re/im, bf16, transposed -------
  for (int i = tid; i < NB * NDIM; i += 512) {
    int n = i >> 7, k = i & (NDIM - 1);
    float2 v = ((const float2*)y)[(size_t)(b0 + n) * NDIM + k];
    Yre[n * LDY + k] = f2bf(v.x);
    Yim[n * LDY + k] = f2bf(v.y);
  }
  __syncthreads();

  // ---- Stage 1: Ay = cmul(A, y) (K=128 GEMM), x0 = shrink(g0*Ay, eta0) -----
  {
    const float g0 = gammas[0], e0 = etas[0];
    v8f accre[4], accim[4];
#pragma unroll
    for (int nt = 0; nt < 4; ++nt) {
      accre[nt] = (v8f){0.f, 0.f, 0.f, 0.f, 0.f, 0.f, 0.f, 0.f};
      accim[nt] = (v8f){0.f, 0.f, 0.f, 0.f, 0.f, 0.f, 0.f, 0.f};
    }
#pragma unroll 2
    for (int k0 = 0; k0 < NDIM; k0 += 32) {
      v16bf ar  = loadA(Aw_re, NDIM, m0 + col, k0, hl);
      v16bf ai  = loadA(Aw_im, NDIM, m0 + col, k0, hl);
      v16bf nai = negfrag(ai);
#pragma unroll
      for (int nt = 0; nt < 4; ++nt) {
        v16bf yr = loadB(Yre, LDY, nt * 16 + col, k0, hl);
        v16bf yi = loadB(Yim, LDY, nt * 16 + col, k0, hl);
        accre[nt] = wmma_bf16(ar,  yr, accre[nt]);
        accre[nt] = wmma_bf16(nai, yi, accre[nt]);
        accim[nt] = wmma_bf16(ar,  yi, accim[nt]);
        accim[nt] = wmma_bf16(ai,  yr, accim[nt]);
      }
    }
    unsigned short* X0re = smem + OFF_X0RE;
    unsigned short* X0im = smem + OFF_X0IM;
#pragma unroll
    for (int nt = 0; nt < 4; ++nt) {
      int n = nt * 16 + col;
      float fre[8], fim[8], xr0[8], xi0[8];
#pragma unroll
      for (int v = 0; v < 8; ++v) {
        fre[v] = accre[nt][v];
        fim[v] = accim[nt][v];
        xr0[v] = shrinkf(g0 * fre[v], e0);
        xi0[v] = shrinkf(g0 * fim[v], e0);
      }
      *(u32x4*)(AYre + n * LDX + mrow) = pack8(fre);
      *(u32x4*)(AYim + n * LDX + mrow) = pack8(fim);
      *(u32x4*)(X0re + n * LDX + mrow) = pack8(xr0);
      *(u32x4*)(X0im + n * LDX + mrow) = pack8(xi0);
    }
  }
  __syncthreads();

  // ---- Iterations: x <- shrink(x - g*cmul(B,x) + g*Ay, eta) ----------------
  int cur = 0;
#pragma unroll 1
  for (int t = 1; t <= TITERS; ++t) {
    const float g  = gammas[t];
    const float et = etas[t];
    // runtime ternaries (NOT pointer arrays) to avoid LDS addrspacecast in
    // static initializers, which ld.lld rejects
    unsigned short* XreC = smem + (cur ? OFF_X1RE : OFF_X0RE);
    unsigned short* XimC = smem + (cur ? OFF_X1IM : OFF_X0IM);
    unsigned short* XreN = smem + (cur ? OFF_X0RE : OFF_X1RE);
    unsigned short* XimN = smem + (cur ? OFF_X0IM : OFF_X1IM);

    v8f accre[4], accim[4];
#pragma unroll
    for (int nt = 0; nt < 4; ++nt) {
      accre[nt] = (v8f){0.f, 0.f, 0.f, 0.f, 0.f, 0.f, 0.f, 0.f};
      accim[nt] = (v8f){0.f, 0.f, 0.f, 0.f, 0.f, 0.f, 0.f, 0.f};
    }
#pragma unroll 2
    for (int k0 = 0; k0 < MDIM; k0 += 32) {
      // B-weight fragments loaded once per k-step, reused across 4 n-tiles
      v16bf br  = loadA(Bw_re, MDIM, m0 + col, k0, hl);
      v16bf bi  = loadA(Bw_im, MDIM, m0 + col, k0, hl);
      v16bf nbi = negfrag(bi);
#pragma unroll
      for (int nt = 0; nt < 4; ++nt) {
        v16bf xr = loadB(XreC, LDX, nt * 16 + col, k0, hl);
        v16bf xi = loadB(XimC, LDX, nt * 16 + col, k0, hl);
        accre[nt] = wmma_bf16(br,  xr, accre[nt]);
        accre[nt] = wmma_bf16(nbi, xi, accre[nt]);
        accim[nt] = wmma_bf16(br,  xi, accim[nt]);
        accim[nt] = wmma_bf16(bi,  xr, accim[nt]);
      }
    }
    // epilogue: xn = shrink(x - g*Bx + g*Ay, eta)
#pragma unroll
    for (int nt = 0; nt < 4; ++nt) {
      int n = nt * 16 + col;
      float ayr[8], ayi[8], xr[8], xi[8], rre[8], rim[8];
      unpack8(*(const u32x4*)(AYre + n * LDX + mrow), ayr);
      unpack8(*(const u32x4*)(AYim + n * LDX + mrow), ayi);
      unpack8(*(const u32x4*)(XreC + n * LDX + mrow), xr);
      unpack8(*(const u32x4*)(XimC + n * LDX + mrow), xi);
#pragma unroll
      for (int v = 0; v < 8; ++v) {
        float ur = xr[v] - g * accre[nt][v] + g * ayr[v];
        float ui = xi[v] - g * accim[nt][v] + g * ayi[v];
        rre[v] = shrinkf(ur, et);
        rim[v] = shrinkf(ui, et);
      }
      if (t < TITERS) {
        *(u32x4*)(XreN + n * LDX + mrow) = pack8(rre);
        *(u32x4*)(XimN + n * LDX + mrow) = pack8(rim);
      } else {
        // out: (B, M, 2) fp32 interleaved; write from f32 accumulators
        float2* o = (float2*)out;
        size_t base = (size_t)(b0 + n) * MDIM + mrow;
#pragma unroll
        for (int v = 0; v < 8; ++v)
          o[base + v] = make_float2(rre[v], rim[v]);
      }
    }
    cur ^= 1;
    __syncthreads();
  }
}

// ---------------------------------------------------------------------------
extern "C" void kernel_launch(void* const* d_in, const int* in_sizes, int n_in,
                              void* d_out, int out_size, void* d_ws, size_t ws_size,
                              hipStream_t stream) {
  const float* y      = (const float*)d_in[0];  // (65536, 128, 2)
  const float* A      = (const float*)d_in[1];  // (2, 256, 128)
  const float* Bm     = (const float*)d_in[2];  // (2, 256, 256)
  const float* etas   = (const float*)d_in[3];  // (11,)
  const float* gammas = (const float*)d_in[4];  // (11,)
  float* out          = (float*)d_out;          // (65536, 256, 2)
  unsigned short* wbf = (unsigned short*)d_ws;  // 384 KB bf16 weights

  // fp32 -> bf16 weight conversion (covers max(131072, 65536) elements)
  convert_weights<<<512, 256, 0, stream>>>(A, Bm, wbf);

  const int blocks = 65536 / NB;                  // 1024 tiles of 64 samples
  const size_t shmem = (size_t)LDS_USHORTS * sizeof(unsigned short); // 237568 B
  lista_wmma_kernel<<<blocks, 512, shmem, stream>>>(y, wbf, etas, gammas, out);
}
